// PtrNet2_26972394619304
// MI455X (gfx1250) — matrix operand order repaired and checked by reading, not compiled, not run on hardware
//
#include <hip/hip_runtime.h>
#include <hip/hip_bf16.h>
#include <math.h>

typedef __attribute__((ext_vector_type(16))) _Float16 v16h;
typedef __attribute__((ext_vector_type(8)))  _Float16 v8h;
typedef __attribute__((ext_vector_type(8)))  float    v8f;

#define BB 512
#define TT 2048
#define HH 128
#define NWAVE 4          // 128 threads per block
#define TILES (TT / 16)  // 128 token tiles

__device__ __forceinline__ v8f wmma_f16(v16h a, v16h b, v8f c) {
  // D = A(16x32,f16) * B(32x16,f16) + C(16x16,f32)
  return __builtin_amdgcn_wmma_f32_16x16x32_f16(
      /*neg_a=*/false, a, /*neg_b=*/false, b,
      /*c_mod=*/(short)0, c, /*reuse_a=*/false, /*reuse_b=*/false);
}

__device__ __forceinline__ float tanh_fast(float x) {
  float ax = fabsf(x);
  float e  = __expf(-2.0f * ax);
  float r  = (1.0f - e) / (1.0f + e);
  return copysignf(r, x);
}

__device__ __forceinline__ v16h loadFrag(const _Float16* frag, int nt, int kc, int lane) {
  const _Float16* p = frag + (((((nt << 2) + kc) << 5) + lane) << 4);
  v8h lo = *(const v8h*)p;
  v8h hi = *(const v8h*)(p + 8);
  return __builtin_shufflevector(lo, hi, 0,1,2,3,4,5,6,7,8,9,10,11,12,13,14,15);
}

__global__ void __launch_bounds__(128, 1)
ptrnet_fused(const float* __restrict__ x,     const float* __restrict__ mask,
             const float* __restrict__ embW,  const float* __restrict__ embB,
             const float* __restrict__ encW,  const float* __restrict__ encB,
             const float* __restrict__ decIn, const float* __restrict__ Vec,
             const float* __restrict__ WqW,   const float* __restrict__ WqB,
             const float* __restrict__ WrefW, const float* __restrict__ WrefB,
             const float* __restrict__ fc1,   const float* __restrict__ fc2,
             float* __restrict__ out) {
  // ---- LDS ----
  __shared__ __align__(16) _Float16 encFrag[16384];            // 32 KB  B-fragments of encW^T
  __shared__ __align__(16) _Float16 wrefFrag[16384];           // 32 KB  B-fragments of WrefW^T
  __shared__ __align__(16) _Float16 refStage[NWAVE][16 * HH];  // 16 KB  per-wave masked ref tile (f16)
  __shared__ float numBuf[NWAVE][HH];                          // per-wave softmax-weighted ref accum
  __shared__ float mzBuf[NWAVE][2];                            // per-wave (max, Z)
  __shared__ float maskbuf[NWAVE][16];
  __shared__ float u1s[HH], querys[HH], abar[HH], hidden[HH];
  __shared__ float embWs[2 * HH], embBs[HH], encBs[HH], wrefBs[HH], Vecs[HH];

  const int tid  = threadIdx.x;
  const int lane = tid & 31;
  const int wv   = tid >> 5;
  const int half = lane >> 4;
  const int ln   = lane & 15;  // row (A) or col (B/C/D) index inside the 16-group
  const int b    = blockIdx.x;

  // ---- Stage constants + convert weights into WMMA B-fragment layout ----
  for (int i = tid; i < 2 * HH; i += 128) embWs[i] = embW[i];
  embBs[tid]  = embB[tid];
  encBs[tid]  = encB[tid];
  wrefBs[tid] = WrefB[tid];
  Vecs[tid]   = Vec[tid];
  querys[tid] = decIn[tid];

  // B[k][n] = W[nt*16+n][k]; fragment slot = ((nt*4+kc)*32 + lane2), 16 halves at k = kc*32 + 8*(lane2>>4) + {0..7, 16..23}
  for (int slot = tid; slot < 1024; slot += 128) {
    int lane2   = slot & 31;
    int kc      = (slot >> 5) & 3;
    int nt      = slot >> 7;
    int row     = nt * 16 + (lane2 & 15);
    int colbase = kc * 32 + ((lane2 >> 4) << 3);
    const float* s1 = encW  + row * HH + colbase;
    const float* s2 = WrefW + row * HH + colbase;
    _Float16* d1 = &encFrag[slot * 16];
    _Float16* d2 = &wrefFrag[slot * 16];
#pragma unroll
    for (int j = 0; j < 16; ++j) { d1[j] = (_Float16)s1[j + (j & 8)]; d2[j] = (_Float16)s2[j + (j & 8)]; }
  }
  __syncthreads();

  // ================= two glimpses =================
  for (int g = 0; g < 2; ++g) {
    // u1 = Wq @ query + Wq_b ; zero per-wave accumulators
    {
      float acc = WqB[tid];
      const float* wr = WqW + tid * HH;
      for (int i = 0; i < HH; ++i) acc += wr[i] * querys[i];
      u1s[tid] = acc;
#pragma unroll
      for (int ww = 0; ww < NWAVE; ++ww) numBuf[ww][tid] = 0.0f;
    }
    __syncthreads();

    float m_run = -3.0e38f, Z_run = 0.0f;

    for (int tile = wv; tile < TILES; tile += NWAVE) {
      const int t0 = tile * 16;
      if (lane < 16) maskbuf[wv][lane] = mask[(size_t)b * TT + t0 + lane];

      // x for this lane's A-row
      const float* xp = x + ((size_t)b * TT + t0 + ln) * 2;
      const float x0 = xp[0], x1 = xp[1];

      // A fragments of x_embed (16x128 f16), built in-register
      v16h a[4];
#pragma unroll
      for (int kc = 0; kc < 4; ++kc) {
#pragma unroll
        for (int e = 0; e < 16; ++e) {
          int r = e >> 1;
          int k = kc * 32 + ((r < 4) ? (2 * r) : (16 + 2 * (r - 4))) + 8 * half + (e & 1);
          float v = x0 * embWs[2 * k] + x1 * embWs[2 * k + 1] + embBs[k];
          a[kc][e] = (_Float16)v;
        }
      }

      // ref = x_embed @ encW^T + enc_b, then mask, store tile to LDS as f16
      for (int nt = 0; nt < 8; ++nt) {
        float bias = encBs[nt * 16 + ln];
        v8f c;
#pragma unroll
        for (int i = 0; i < 8; ++i) c[i] = bias;
#pragma unroll
        for (int kc = 0; kc < 4; ++kc) c = wmma_f16(a[kc], loadFrag(encFrag, nt, kc, lane), c);
#pragma unroll
        for (int r = 0; r < 8; ++r) {
          int trow = r + 8 * half;
          float rv = c[r] * maskbuf[wv][trow];
          refStage[wv][trow * HH + nt * 16 + ln] = (_Float16)rv;
        }
      }

      // reload masked ref as A fragments
      v16h ra[4];
#pragma unroll
      for (int kc = 0; kc < 4; ++kc) {
        const _Float16* pr = &refStage[wv][ln * HH + kc * 32 + 8 * half];
        v8h lo = *(const v8h*)pr;
        v8h hi = *(const v8h*)(pr + 16);
        ra[kc] = __builtin_shufflevector(lo, hi, 0,1,2,3,4,5,6,7,8,9,10,11,12,13,14,15);
      }

      // u2 = ref @ WrefW^T + Wref_b ; s_t = sum_o Vec[o]*tanh(u1[o]+u2[t,o])
      float p[8];
#pragma unroll
      for (int r = 0; r < 8; ++r) p[r] = 0.0f;
      for (int nt = 0; nt < 8; ++nt) {
        int o = nt * 16 + ln;
        float bias = wrefBs[o];
        v8f c;
#pragma unroll
        for (int i = 0; i < 8; ++i) c[i] = bias;
#pragma unroll
        for (int kc = 0; kc < 4; ++kc) c = wmma_f16(ra[kc], loadFrag(wrefFrag, nt, kc, lane), c);
        float u1o = u1s[o], vo = Vecs[o];
#pragma unroll
        for (int r = 0; r < 8; ++r) p[r] += vo * tanh_fast(u1o + c[r]);
      }
      // reduce over the 16 lanes of each half-group: p[r] -> s[t0 + r + 8*half]
#pragma unroll
      for (int off = 1; off < 16; off <<= 1)
#pragma unroll
        for (int r = 0; r < 8; ++r) p[r] += __shfl_xor(p[r], off, 32);

      // online softmax update (state replicated per lane, consistent per wave)
      float tmax = p[0];
#pragma unroll
      for (int r = 1; r < 8; ++r) tmax = fmaxf(tmax, p[r]);
      tmax = fmaxf(tmax, __shfl_xor(tmax, 16, 32));
      float m_new = fmaxf(m_run, tmax);
      float scale = __expf(m_run - m_new);
      float w[8], zs = 0.0f;
#pragma unroll
      for (int r = 0; r < 8; ++r) { w[r] = __expf(p[r] - m_new); zs += w[r]; }
      zs += __shfl_xor(zs, 16, 32);
      Z_run = Z_run * scale + zs;
      m_run = m_new;

      float wt[16];
#pragma unroll
      for (int r = 0; r < 8; ++r) {
        float wo = __shfl_xor(w[r], 16, 32);
        wt[r]     = (half == 0) ? w[r] : wo;
        wt[8 + r] = (half == 0) ? wo   : w[r];
      }
      // num[h] = num[h]*scale + sum_t wt[t]*ref[t,h]
#pragma unroll
      for (int j = 0; j < 4; ++j) {
        int h = lane + (j << 5);
        float acc = numBuf[wv][h] * scale;
#pragma unroll
        for (int t = 0; t < 16; ++t) acc += wt[t] * (float)refStage[wv][t * HH + h];
        numBuf[wv][h] = acc;
      }
    }

    if (lane == 0) { mzBuf[wv][0] = m_run; mzBuf[wv][1] = Z_run; }
    __syncthreads();

    // merge waves: global softmax, then abar = (sum_t a_t ref_t)
    {
      float gm = fmaxf(fmaxf(mzBuf[0][0], mzBuf[1][0]), fmaxf(mzBuf[2][0], mzBuf[3][0]));
      float gZ = 0.0f, nf = 0.0f;
#pragma unroll
      for (int ww = 0; ww < NWAVE; ++ww) {
        float e = __expf(mzBuf[ww][0] - gm);
        gZ += mzBuf[ww][1] * e;
        nf += numBuf[ww][tid] * e;
      }
      abar[tid] = nf / gZ;
    }
    __syncthreads();

    // query_next = Wref @ abar + Wref_b   (since sum_t a_t = 1)
    {
      float acc = wrefBs[tid];
      const float* wr = WrefW + tid * HH;
      for (int i = 0; i < HH; ++i) acc += wr[i] * abar[i];
      __syncthreads();
      querys[tid] = acc;
    }
    __syncthreads();
  }

  // final2FC: relu(q @ fc1^T) @ fc2^T
  {
    float acc = 0.0f;
    const float* wr = fc1 + tid * HH;
    for (int i = 0; i < HH; ++i) acc += wr[i] * querys[i];
    hidden[tid] = fmaxf(acc, 0.0f);
  }
  __syncthreads();
  if (tid < 2) {
    float acc = 0.0f;
    const float* wr = fc2 + tid * HH;
    for (int h = 0; h < HH; ++h) acc += wr[h] * hidden[h];
    out[b * 2 + tid] = acc;
  }
}

extern "C" void kernel_launch(void* const* d_in, const int* in_sizes, int n_in,
                              void* d_out, int out_size, void* d_ws, size_t ws_size,
                              hipStream_t stream) {
  const float* x     = (const float*)d_in[0];
  const float* mask  = (const float*)d_in[1];
  const float* embW  = (const float*)d_in[2];
  const float* embB  = (const float*)d_in[3];
  const float* encW  = (const float*)d_in[4];
  const float* encB  = (const float*)d_in[5];
  const float* decIn = (const float*)d_in[6];
  const float* Vec   = (const float*)d_in[7];
  const float* WqW   = (const float*)d_in[8];
  const float* WqB   = (const float*)d_in[9];
  const float* WrefW = (const float*)d_in[10];
  const float* WrefB = (const float*)d_in[11];
  const float* fc1   = (const float*)d_in[12];
  const float* fc2   = (const float*)d_in[13];
  float* out = (float*)d_out;

  ptrnet_fused<<<BB, 128, 0, stream>>>(x, mask, embW, embB, encW, encB, decIn, Vec,
                                       WqW, WqB, WrefW, WrefB, fc1, fc2, out);
}
